// PairedKidneyModel_84920093376791
// MI455X (gfx1250) — compile-verified, matrix operand order checked by reading, not run on hardware
//
#include <hip/hip_runtime.h>

// ---------------------------------------------------------------------------
// GAT stack for MI455X (gfx1250): bf16 WMMA GEMMs (v_wmma_f32_16x16x32_bf16)
// with double-buffered LDS tiles staged by GLOBAL_LOAD_ASYNC_TO_LDS_B128
// (ASYNCcnt-tracked, no VGPR round trip), dense masked column-softmax
// producing alpha^T in bf16, fused bias/ReLU epilogues.
// Shapes: N=4096 nodes, H=256 hidden, L=3 layers. All GEMM dims divide the
// tile sizes (BM=128, BN=64, BK=32): no bounds checks, EXEC==all-ones for
// every WMMA (ISA requirement).
// ---------------------------------------------------------------------------

typedef __attribute__((ext_vector_type(16))) __bf16 v16bf;
typedef __attribute__((ext_vector_type(8)))  float  v8f;
typedef __attribute__((ext_vector_type(4)))  int    v4i;

#if defined(__has_builtin)
#  if __has_builtin(__builtin_amdgcn_global_load_async_to_lds_b128) && \
      __has_builtin(__builtin_amdgcn_s_wait_asynccnt)
#    define USE_ASYNC_LDS 1
#  endif
#endif
#ifndef USE_ASYNC_LDS
#  define USE_ASYNC_LDS 0
#endif

#if USE_ASYNC_LDS
// Builtin prototype (from hipcc diagnostic): param0 = v4i addrspace(1)*,
// param1 = v4i addrspace(3)*.  AS3 pointers are 32-bit; the low 32 bits of a
// generic LDS address are the LDS offset (ISA 10.2).
// AS1 shares the generic address value.
typedef __attribute__((address_space(1))) v4i gv4i;
typedef __attribute__((address_space(3))) v4i lv4i;

__device__ __forceinline__ gv4i* to_gbl_v4(const void* p) {
  return (gv4i*)(unsigned long long)p;
}
__device__ __forceinline__ lv4i* to_lds_v4(void* p) {
  return (lv4i*)(unsigned int)(unsigned long long)p;
}
#endif

// ---------------- f32 -> bf16 convert ----------------
__global__ void f2bf_kernel(const float* __restrict__ in, __bf16* __restrict__ out, int n) {
  int i = blockIdx.x * 256 + threadIdx.x;
  if (i < n) out[i] = (__bf16)in[i];
}

// ---------------- stage-1 embedding: [N,2]@[2,H] + b1, emit bf16 ----------------
__global__ void embed1_kernel(const float* __restrict__ arrival,
                              const float* __restrict__ departure,
                              const float* __restrict__ hard,
                              const float* __restrict__ W1,   // [2,H]
                              const float* __restrict__ b1,   // [H]
                              const int*   __restrict__ tstep,
                              __bf16* __restrict__ t1,        // [N,H] bf16
                              int H) {
  int idx = blockIdx.x * 256 + threadIdx.x;     // N*H threads
  int n = idx / H;
  int h = idx - n * H;
  float den = departure[n] - arrival[n];
  den = (den != 0.0f) ? den : 1.0f;
  float prog = ((float)tstep[0] - arrival[n]) / den;
  float v = prog * W1[h] + hard[n] * W1[H + h] + b1[h];
  t1[idx] = (__bf16)v;
}

// ---------------- tiled bf16 WMMA GEMM: C = A[MxK] @ B[KxN] (+bias)(+relu) ----------------
#define BM 128
#define BN 64
#define BK 32
#define LPITCH 40   // shorts per LDS row: keeps 16B alignment, staggers banks

__global__ __launch_bounds__(256)
void gemm_bf16_wmma(const __bf16* __restrict__ A, const __bf16* __restrict__ B,
                    const float* __restrict__ bias,
                    float* __restrict__ Cf, __bf16* __restrict__ Cb,
                    int M, int N, int K, int relu)
{
  __shared__ __bf16 As[2][BM * LPITCH];   // 2 x 10240 B
  __shared__ __bf16 Bs[2][BN * LPITCH];   // 2 x  5120 B, transposed: Bs[n][k]

  const int t    = threadIdx.x;
  const int lane = t & 31;
  const int wave = t >> 5;       // 8 waves
  const int hh   = lane >> 4;    // lane-half (fragment K-half select)
  const int ln   = lane & 15;    // row (A) / col (B,C) within 16x16 tile
  const int wm   = wave & 3;     // 4 waves along M -> 32 rows each
  const int wn   = wave >> 2;    // 2 waves along N -> 32 cols each
  const int m0   = blockIdx.y * BM;
  const int n0   = blockIdx.x * BN;

  // per-thread staging coordinates
  const int arow0 = t >> 2;           // A chunk 0: rows 0..63
  const int acb   = t & 3;            // A chunk column block (8 bf16 each)
  const int bkr   = t >> 3;           // B source row (k) 0..31
  const int bnb   = t & 7;            // B col block of 8

  v8f acc[2][2];
#pragma unroll
  for (int i = 0; i < 2; ++i)
#pragma unroll
    for (int j = 0; j < 2; ++j)
#pragma unroll
      for (int e = 0; e < 8; ++e) acc[i][j][e] = 0.0f;

  // ---- stage one BMxBK A tile + BKxBN B tile into buffer `buf` ----
  auto stage = [&](int buf, int k0) {
#if USE_ASYNC_LDS
    // A tile: direct memory->LDS, 16B per lane, 2 issues per thread.
    __builtin_amdgcn_global_load_async_to_lds_b128(
        to_gbl_v4(A + (size_t)(m0 + arow0) * K + k0 + acb * 8),
        to_lds_v4(&As[buf][arow0 * LPITCH + acb * 8]), 0, 0);
    __builtin_amdgcn_global_load_async_to_lds_b128(
        to_gbl_v4(A + (size_t)(m0 + arow0 + 64) * K + k0 + acb * 8),
        to_lds_v4(&As[buf][(arow0 + 64) * LPITCH + acb * 8]), 0, 0);
#else
#pragma unroll
    for (int rep = 0; rep < 2; ++rep) {
      int row = arow0 + rep * 64;
      uint4 v = *(const uint4*)(A + (size_t)(m0 + row) * K + k0 + acb * 8);
      *(uint4*)(&As[buf][row * LPITCH + acb * 8]) = v;
    }
#endif
    // B tile 32x64, transposed into Bs[buf][n][k] (element transpose -> VGPR path)
    {
      uint4 v = *(const uint4*)(B + (size_t)(k0 + bkr) * N + n0 + bnb * 8);
      const __bf16* sv = (const __bf16*)&v;
#pragma unroll
      for (int j = 0; j < 8; ++j)
        Bs[buf][(bnb * 8 + j) * LPITCH + bkr] = sv[j];
    }
  };

  stage(0, 0);
#if USE_ASYNC_LDS
  __builtin_amdgcn_s_wait_asynccnt(0);
#endif
  __syncthreads();

  int cur = 0;
  for (int k0 = 0; k0 < K; k0 += BK) {
    // issue next tile into the other buffer while we compute this one
    if (k0 + BK < K) {
      stage(cur ^ 1, k0 + BK);
      // keep a prefetch on the B stream (global_prefetch_b8)
      __builtin_prefetch((const void*)(B + (size_t)(k0 + BK + bkr) * N + n0 + bnb * 8), 0, 3);
    }

    // ---- fragments per documented 16-bit A layout:
    // half 0: elems 0..7 -> K 0..7,  elems 8..15 -> K 16..23
    // half 1: elems 0..7 -> K 8..15, elems 8..15 -> K 24..31
    v16bf afr[2], bfr[2];
#pragma unroll
    for (int tm = 0; tm < 2; ++tm) {
      const __bf16* p = &As[cur][(wm * 32 + tm * 16 + ln) * LPITCH];
      uint4* d = (uint4*)&afr[tm];
      d[0] = *(const uint4*)(p + 8 * hh);
      d[1] = *(const uint4*)(p + 16 + 8 * hh);
    }
#pragma unroll
    for (int tn = 0; tn < 2; ++tn) {
      const __bf16* p = &Bs[cur][(wn * 32 + tn * 16 + ln) * LPITCH];
      uint4* d = (uint4*)&bfr[tn];
      d[0] = *(const uint4*)(p + 8 * hh);
      d[1] = *(const uint4*)(p + 16 + 8 * hh);
    }
#pragma unroll
    for (int tm = 0; tm < 2; ++tm)
#pragma unroll
      for (int tn = 0; tn < 2; ++tn)
        acc[tm][tn] = __builtin_amdgcn_wmma_f32_16x16x32_bf16(
            false, afr[tm], false, bfr[tn], (short)0, acc[tm][tn], false, false);

#if USE_ASYNC_LDS
    __builtin_amdgcn_s_wait_asynccnt(0);   // next-tile async copies landed
#endif
    __syncthreads();                        // waits dscnt, then barrier
    cur ^= 1;
  }

  // ---- epilogue: C/D layout -> row = r + 8*half, col = lane&15 ----
#pragma unroll
  for (int tm = 0; tm < 2; ++tm)
#pragma unroll
    for (int tn = 0; tn < 2; ++tn) {
      int colg = n0 + wn * 32 + tn * 16 + ln;
      float bv = bias ? bias[colg] : 0.0f;
#pragma unroll
      for (int r = 0; r < 8; ++r) {
        int rowg = m0 + wm * 32 + tm * 16 + r + 8 * hh;
        float v = acc[tm][tn][r] + bv;
        if (relu) v = fmaxf(v, 0.0f);
        size_t idx = (size_t)rowg * N + colg;
        Cf[idx] = v;
        Cb[idx] = (__bf16)v;
      }
    }
}

// ---------------- per-node attention scores: a_s = hw@asrc, a_d = hw@adst ----------------
__global__ __launch_bounds__(256)
void rowdot2_kernel(const float* __restrict__ hw,
                    const float* __restrict__ asrc, const float* __restrict__ adst,
                    float* __restrict__ a_s, float* __restrict__ a_d, int H) {
  int n = blockIdx.x;
  int t = threadIdx.x;           // == H == 256
  __shared__ float s1[256];
  __shared__ float s2[256];
  float v = hw[(size_t)n * H + t];
  s1[t] = v * asrc[t];
  s2[t] = v * adst[t];
  __syncthreads();
  for (int off = 128; off > 0; off >>= 1) {
    if (t < off) { s1[t] += s1[t + off]; s2[t] += s2[t + off]; }
    __syncthreads();
  }
  if (t == 0) { a_s[n] = s1[0]; a_d[n] = s2[0]; }
}

// ---------------- masked column softmax -> alphaT[j,:] in bf16 ----------------
__global__ __launch_bounds__(256)
void softmax_col_kernel(const float* __restrict__ adj,
                        const unsigned char* __restrict__ mask,
                        const float* __restrict__ a_s, const float* __restrict__ a_d,
                        __bf16* __restrict__ alphaT, int Nn) {
  int j = blockIdx.x;            // destination node (one alphaT row)
  int t = threadIdx.x;
  __shared__ float red[256];
  const bool  mj = mask[j] != 0;
  const float ad = a_d[j];

  // pass 1: column max of leaky_relu(a_s[i]+a_d[j]) over edge-mask
  float mx = -1e30f;
  for (int i = t; i < Nn; i += 256) {
    bool em = (i == j) || (mj && (mask[i] != 0) && (adj[(size_t)i * Nn + j] != 0.0f));
    if (em) {
      float e = a_s[i] + ad;
      e = (e > 0.0f) ? e : 0.2f * e;
      mx = fmaxf(mx, e);
    }
  }
  red[t] = mx; __syncthreads();
  for (int off = 128; off > 0; off >>= 1) {
    if (t < off) red[t] = fmaxf(red[t], red[t + off]);
    __syncthreads();
  }
  mx = red[0]; __syncthreads();

  // pass 2: column sum of exp(e - mx)
  float s = 0.0f;
  for (int i = t; i < Nn; i += 256) {
    bool em = (i == j) || (mj && (mask[i] != 0) && (adj[(size_t)i * Nn + j] != 0.0f));
    if (em) {
      float e = a_s[i] + ad;
      e = (e > 0.0f) ? e : 0.2f * e;
      s += __expf(e - mx);
    }
  }
  red[t] = s; __syncthreads();
  for (int off = 128; off > 0; off >>= 1) {
    if (t < off) red[t] += red[t + off];
    __syncthreads();
  }
  const float inv = 1.0f / red[0];
  __syncthreads();

  // pass 3: write alphaT[j,i] (coalesced in i)
  for (int i = t; i < Nn; i += 256) {
    bool em = (i == j) || (mj && (mask[i] != 0) && (adj[(size_t)i * Nn + j] != 0.0f));
    float a = 0.0f;
    if (em) {
      float e = a_s[i] + ad;
      e = (e > 0.0f) ? e : 0.2f * e;
      a = __expf(e - mx) * inv;
    }
    alphaT[(size_t)j * Nn + i] = (__bf16)a;
  }
}

// ---------------- residual + LayerNorm + selector head + sigmoid + mask ----------------
__global__ __launch_bounds__(256)
void head_kernel(const float* __restrict__ x, const float* __restrict__ h,
                 const unsigned char* __restrict__ mask,
                 const float* __restrict__ selW, const float* __restrict__ selb,
                 const int* __restrict__ tstep, const int* __restrict__ ttot,
                 float* __restrict__ out, int H) {
  int n = blockIdx.x;
  int t = threadIdx.x;           // == H == 256
  __shared__ float red[256];
  float v = x[(size_t)n * H + t] + h[(size_t)n * H + t];

  red[t] = v; __syncthreads();
  for (int off = 128; off > 0; off >>= 1) { if (t < off) red[t] += red[t + off]; __syncthreads(); }
  float mu = red[0] * (1.0f / 256.0f); __syncthreads();

  float d = v - mu;
  red[t] = d * d; __syncthreads();
  for (int off = 128; off > 0; off >>= 1) { if (t < off) red[t] += red[t + off]; __syncthreads(); }
  float var = red[0] * (1.0f / 256.0f); __syncthreads();

  float xn = d * rsqrtf(var + 1e-5f);
  red[t] = xn * selW[t]; __syncthreads();
  for (int off = 128; off > 0; off >>= 1) { if (t < off) red[t] += red[t + off]; __syncthreads(); }

  if (t == 0) {
    float tctx  = (float)tstep[0] / (float)ttot[0];
    float logit = red[0] + tctx * selW[H] + selb[0];
    float sig   = 1.0f / (1.0f + __expf(-logit));
    out[n] = (mask[n] != 0) ? sig : 0.0f;
  }
}

// ---------------------------------------------------------------------------
extern "C" void kernel_launch(void* const* d_in, const int* in_sizes, int n_in,
                              void* d_out, int out_size, void* d_ws, size_t ws_size,
                              hipStream_t stream) {
  const float* adj       = (const float*)d_in[0];
  const float* arrival   = (const float*)d_in[1];
  const float* departure = (const float*)d_in[2];
  const float* hard      = (const float*)d_in[3];
  const unsigned char* mask = (const unsigned char*)d_in[4];
  const float* embW1 = (const float*)d_in[5];
  const float* embb1 = (const float*)d_in[6];
  const float* embW2 = (const float*)d_in[7];
  const float* embb2 = (const float*)d_in[8];
  const float* gatW  = (const float*)d_in[9];
  const float* gatAs = (const float*)d_in[10];
  const float* gatAd = (const float*)d_in[11];
  const float* gatB  = (const float*)d_in[12];
  const float* selW  = (const float*)d_in[13];
  const float* selb  = (const float*)d_in[14];
  const int*   tstep = (const int*)d_in[15];
  const int*   ttot  = (const int*)d_in[16];
  float* out = (float*)d_out;

  const int Nn = in_sizes[1];                 // 4096
  const int H  = in_sizes[6];                 // 256
  const int L  = in_sizes[12] / H;            // 3
  (void)n_in; (void)out_size; (void)ws_size; (void)embW1;

  // ---- workspace layout (256B aligned) ----
  char* ws = (char*)d_ws;
  size_t off = 0;
  auto alloc = [&](size_t bytes) -> void* {
    void* p = ws + off;
    off += (bytes + 255) & ~(size_t)255;
    return p;
  };
  __bf16* t1_bf   = (__bf16*)alloc((size_t)Nn * H * 2);
  float*  x_f     = (float*) alloc((size_t)Nn * H * 4);
  __bf16* h_bf    = (__bf16*)alloc((size_t)Nn * H * 2);
  float*  hw_f    = (float*) alloc((size_t)Nn * H * 4);
  __bf16* hw_bf   = (__bf16*)alloc((size_t)Nn * H * 2);
  float*  h_f     = (float*) alloc((size_t)Nn * H * 4);
  float*  a_s     = (float*) alloc((size_t)Nn * 4);
  float*  a_d     = (float*) alloc((size_t)Nn * 4);
  __bf16* embW2bf = (__bf16*)alloc((size_t)H * H * 2);
  __bf16* gatWbf  = (__bf16*)alloc((size_t)L * H * H * 2);
  __bf16* alphaT  = (__bf16*)alloc((size_t)Nn * Nn * 2);

  // ---- weight conversion to bf16 ----
  f2bf_kernel<<<(H * H + 255) / 256, 256, 0, stream>>>(embW2, embW2bf, H * H);
  f2bf_kernel<<<(L * H * H + 255) / 256, 256, 0, stream>>>(gatW, gatWbf, L * H * H);

  // ---- features + stage-1 embedding (K=2) ----
  embed1_kernel<<<(Nn * H) / 256, 256, 0, stream>>>(arrival, departure, hard,
                                                    embW1, embb1, tstep, t1_bf, H);

  dim3 blk(256);
  dim3 grid_small(H / BN, Nn / BM);   // (4, 32)

  // ---- x = t1 @ embW2 + b2 ; also emit bf16 copy as layer-0 input h ----
  gemm_bf16_wmma<<<grid_small, blk, 0, stream>>>(t1_bf, embW2bf, embb2,
                                                 x_f, h_bf, Nn, H, H, 0);

  // ---- GAT layers ----
  for (int l = 0; l < L; ++l) {
    // hw = h @ W_l
    gemm_bf16_wmma<<<grid_small, blk, 0, stream>>>(h_bf, gatWbf + (size_t)l * H * H,
                                                   nullptr, hw_f, hw_bf, Nn, H, H, 0);
    // a_s, a_d
    rowdot2_kernel<<<Nn, blk, 0, stream>>>(hw_f, gatAs + (size_t)l * H,
                                           gatAd + (size_t)l * H, a_s, a_d, H);
    // alphaT (masked column softmax)
    softmax_col_kernel<<<Nn, blk, 0, stream>>>(adj, mask, a_s, a_d, alphaT, Nn);
    // h = alphaT @ hw + bias_l (+ relu except last layer)
    gemm_bf16_wmma<<<grid_small, blk, 0, stream>>>(alphaT, hw_bf,
                                                   gatB + (size_t)l * H,
                                                   h_f, h_bf, Nn, H, Nn, (l < L - 1) ? 1 : 0);
  }

  // ---- residual + LN + head ----
  head_kernel<<<Nn, blk, 0, stream>>>(x_f, h_f, mask, selW, selb, tstep, ttot, out, H);
}
